// MultiScaleGrid_65541200937476
// MI455X (gfx1250) — matrix operand order repaired and checked by reading, not compiled, not run on hardware
//
#include <hip/hip_runtime.h>
#include <hip/hip_bf16.h>

typedef _Float16 v16h __attribute__((ext_vector_type(16)));
typedef _Float16 h8   __attribute__((ext_vector_type(8)));
typedef float    v8f  __attribute__((ext_vector_type(8)));

#define N_RAYS   1024
#define N_SAMP   891
#define NPTS     (N_RAYS * N_SAMP)          // 912384 = 3564 * 256 exactly
#define RES3     2097152                    // 128^3
#define FSTRIDE  72                         // padded LDS row stride (halves), 144B (16B-mult)
#define ACT_SHIFT (-4.5951198501345889f)    // log(1/(1-0.01)-1)

// load a 16-half WMMA fragment from two contiguous 8-half groups
__device__ __forceinline__ v16h ldfrag(const _Float16* p0, const _Float16* p1) {
  h8 a = *(const h8*)p0;
  h8 b = *(const h8*)p1;
  v16h r;
#pragma unroll
  for (int i = 0; i < 8; i++) { r[i] = a[i]; r[i + 8] = b[i]; }
  return r;
}

// ---------------------------------------------------------------------------
// Kernel 1: 5x5x5 sobel conv over density channel -> normalized normals grid
// ---------------------------------------------------------------------------
__global__ __launch_bounds__(256) void sobel_normals_kernel(
    const float* __restrict__ grid, const float* __restrict__ sk,
    float* __restrict__ normals) {
  __shared__ float k[375];
  for (int i = threadIdx.x; i < 375; i += 256) k[i] = sk[i];
  __syncthreads();
  int id = blockIdx.x * 256 + threadIdx.x;    // < 128^3
  int ic = id & 127;
  int ib = (id >> 7) & 127;
  int ia = id >> 14;
  float acc0 = 0.f, acc1 = 0.f, acc2 = 0.f;
  for (int da = 0; da < 5; da++) {
    int A = ia + da - 2; if ((unsigned)A > 127u) continue;
    for (int db = 0; db < 5; db++) {
      int B = ib + db - 2; if ((unsigned)B > 127u) continue;
      int rowbase = (A * 128 + B) * 128;
      int kbase = (da * 5 + db) * 5;
      for (int dc = 0; dc < 5; dc++) {
        int C = ic + dc - 2; if ((unsigned)C > 127u) continue;
        float d = grid[rowbase + C];
        int ki = kbase + dc;
        acc0 += d * k[ki];
        acc1 += d * k[125 + ki];
        acc2 += d * k[250 + ki];
      }
    }
  }
  float nn  = sqrtf(acc0 * acc0 + acc1 * acc1 + acc2 * acc2);
  float inv = -1.0f / fmaxf(nn, 1e-12f);
  normals[id]            = acc0 * inv;
  normals[RES3 + id]     = acc1 * inv;
  normals[2 * RES3 + id] = acc2 * inv;
}

// ---------------------------------------------------------------------------
// One MLP layer on WMMA: 16x64 tile, in-place through LDS stage (stride 72)
// ---------------------------------------------------------------------------
__device__ __forceinline__ void mlp_layer(v16h a0, v16h a1,
                                          const _Float16* wt /*[64][64] row=n*/,
                                          const float* __restrict__ bias,
                                          _Float16* stg /*16 x FSTRIDE*/,
                                          int lane) {
  int m = lane & 15, hi = lane >> 4;
#pragma unroll
  for (int nt = 0; nt < 4; nt++) {
    float bv = bias[nt * 16 + m];
    v8f c = {bv, bv, bv, bv, bv, bv, bv, bv};
    const _Float16* p0 = wt + (nt * 16 + m) * 64 + hi * 8;    // ktile 0
    v16h bf0 = ldfrag(p0, p0 + 16);
    c = __builtin_amdgcn_wmma_f32_16x16x32_f16(false, a0, false, bf0, (short)0, c,
                                               false, false);
    const _Float16* p1 = p0 + 32;                             // ktile 1
    v16h bf1 = ldfrag(p1, p1 + 16);
    c = __builtin_amdgcn_wmma_f32_16x16x32_f16(false, a1, false, bf1, (short)0, c,
                                               false, false);
#pragma unroll
    for (int r2 = 0; r2 < 8; r2++) {
      float v = fmaxf(c[r2], 0.0f);                            // ReLU
      stg[(r2 + hi * 8) * FSTRIDE + nt * 16 + m] = (_Float16)v;
    }
  }
}

// ---------------------------------------------------------------------------
// Kernel 2 (fused): per-point sampling + feature build in LDS + WMMA MLP.
// point index p = s*1024 + r (sample-major). 256 pts/block; 8 waves x 2 tiles.
// ---------------------------------------------------------------------------
__global__ __launch_bounds__(256) void fused_sample_mlp_kernel(
    const float* __restrict__ ro, const float* __restrict__ rd,
    const float* __restrict__ vdirs, const float* __restrict__ grid,
    const float* __restrict__ normals,
    const float* __restrict__ w0, const float* __restrict__ b0,
    const float* __restrict__ w1, const float* __restrict__ b1,
    const float* __restrict__ w2, const float* __restrict__ b2,
    const float* __restrict__ w3, const float* __restrict__ b3,
    float* __restrict__ alpha, float* __restrict__ rgb /* 3 planes of NPTS */) {
  __shared__ _Float16 sWT[3][64][64];     // transposed weights [layer][n][k]
  __shared__ _Float16 sW3T[16][64];       // output layer, N padded 3->16
  __shared__ _Float16 sFeat[256][FSTRIDE];// feature rows; reused as hidden stage
  __shared__ unsigned char sMask[256];

  int tid = threadIdx.x;
  // ---- weight staging (f16, transposed) ----
  for (int i = tid; i < 3 * 4096; i += 256) {
    int l = i >> 12, rem = i & 4095, n = rem >> 6, k = rem & 63;
    float v;
    if (l == 0)      v = (k < 54) ? w0[k * 64 + n] : 0.0f;
    else if (l == 1) v = w1[k * 64 + n];
    else             v = w2[k * 64 + n];
    sWT[l][n][k] = (_Float16)v;
  }
  for (int i = tid; i < 1024; i += 256) {
    int n = i >> 6, k = i & 63;
    sW3T[n][k] = (_Float16)((n < 3) ? w3[k * 3 + n] : 0.0f);
  }

  // ---- per-thread sampling phase ----
  int p = blockIdx.x * 256 + tid;           // grid sized so p < NPTS always
  int r = p & (N_RAYS - 1);
  int s = p >> 10;

  float ox = ro[3 * r], oy = ro[3 * r + 1], oz = ro[3 * r + 2];
  float dx = rd[3 * r], dy = rd[3 * r + 1], dz = rd[3 * r + 2];
  float vx = (dx == 0.0f) ? 1e-6f : dx;
  float vy = (dy == 0.0f) ? 1e-6f : dy;
  float vz = (dz == 0.0f) ? 1e-6f : dz;
  float rax = (1.0f - ox) / vx, rbx = (-1.0f - ox) / vx;
  float ray_ = (1.0f - oy) / vy, rby = (-1.0f - oy) / vy;
  float raz = (1.0f - oz) / vz, rbz = (-1.0f - oz) / vz;
  float tmin = fmaxf(fmaxf(fminf(rax, rbx), fminf(ray_, rby)), fminf(raz, rbz));
  float tmax = fminf(fminf(fmaxf(rax, rbx), fmaxf(ray_, rby)), fmaxf(raz, rbz));
  tmin = fminf(fmaxf(tmin, 0.2f), 3.0f);
  tmax = fminf(fmaxf(tmax, 0.2f), 3.0f);
  bool rayok = !(tmax <= tmin);

  float dn = sqrtf(dx * dx + dy * dy + dz * dz);
  float t  = tmin + (0.5f * (2.0f / 128.0f)) * (float)s / dn;
  float px = ox + dx * t, py = oy + dy * t, pz = oz + dz * t;
  bool inbox = rayok && px >= -1.0f && px <= 1.0f && py >= -1.0f &&
               py <= 1.0f && pz >= -1.0f && pz <= 1.0f;

  float lat[16];
#pragma unroll
  for (int c = 0; c < 16; c++) lat[c] = 0.0f;
  float nsx = 0.f, nsy = 0.f, nsz = 0.f;

  if (inbox) {
    float ua = (px + 1.0f) * 0.5f * 127.0f;   // stride 128*128 dim
    float ub = (py + 1.0f) * 0.5f * 127.0f;   // stride 128 dim
    float uc = (pz + 1.0f) * 0.5f * 127.0f;   // stride 1 dim
    float fa0 = floorf(ua), fb0 = floorf(ub), fc0 = floorf(uc);
    float fa = ua - fa0, fb = ub - fb0, fc = uc - fc0;
    int a0i = min(max((int)fa0, 0), 127);
    int b0i = min(max((int)fb0, 0), 127);
    int c0i = min(max((int)fc0, 0), 127);
    int a1i = min(a0i + 1, 127), b1i = min(b0i + 1, 127), c1i = min(c0i + 1, 127);
    int i000 = (a0i * 128 + b0i) * 128 + c0i;
    int i001 = (a0i * 128 + b0i) * 128 + c1i;
    int i010 = (a0i * 128 + b1i) * 128 + c0i;
    int i011 = (a0i * 128 + b1i) * 128 + c1i;
    int i100 = (a1i * 128 + b0i) * 128 + c0i;
    int i101 = (a1i * 128 + b0i) * 128 + c1i;
    int i110 = (a1i * 128 + b1i) * 128 + c0i;
    int i111 = (a1i * 128 + b1i) * 128 + c1i;
    float w000 = (1 - fa) * (1 - fb) * (1 - fc), w001 = (1 - fa) * (1 - fb) * fc;
    float w010 = (1 - fa) * fb * (1 - fc),       w011 = (1 - fa) * fb * fc;
    float w100 = fa * (1 - fb) * (1 - fc),       w101 = fa * (1 - fb) * fc;
    float w110 = fa * fb * (1 - fc),             w111 = fa * fb * fc;
#pragma unroll 4
    for (int c = 0; c < 16; c++) {
      const float* g = grid + (size_t)c * RES3;
      lat[c] = g[i000] * w000 + g[i001] * w001 + g[i010] * w010 + g[i011] * w011 +
               g[i100] * w100 + g[i101] * w101 + g[i110] * w110 + g[i111] * w111;
    }
    float ns[3];
#pragma unroll
    for (int c = 0; c < 3; c++) {
      const float* g = normals + (size_t)c * RES3;
      ns[c] = g[i000] * w000 + g[i001] * w001 + g[i010] * w010 + g[i011] * w011 +
              g[i100] * w100 + g[i101] * w101 + g[i110] * w110 + g[i111] * w111;
    }
    float nn  = sqrtf(ns[0] * ns[0] + ns[1] * ns[1] + ns[2] * ns[2]);
    float inv = -1.0f / fmaxf(nn, 1e-12f);
    nsx = ns[0] * inv; nsy = ns[1] * inv; nsz = ns[2] * inv;
  }

  float a_out = 0.0f;
  if (inbox) {
    float d0 = lat[0] + ACT_SHIFT;
    float sp = (d0 > 20.0f) ? d0 : log1pf(__expf(d0));
    a_out = 1.0f - __expf(-sp * 0.5f);
  }
  alpha[p]   = a_out;
  sMask[tid] = inbox ? (unsigned char)1 : (unsigned char)0;

  float wx = vdirs[3 * r], wy = vdirs[3 * r + 1], wz = vdirs[3 * r + 2];
  float dot = -(wx * nsx + wy * nsy + wz * nsz);
  float rx = 2.0f * dot * nsx + wx;
  float ry = 2.0f * dot * nsy + wy;
  float rz = 2.0f * dot * nsz + wz;

  float f[64];
#pragma unroll
  for (int c = 0; c < 15; c++) f[c] = inbox ? lat[c + 1] : 0.0f;
  f[15] = rx; f[16] = ry; f[17] = rz;
  float fr = 1.0f;
#pragma unroll
  for (int j = 0; j < 6; j++) {
    f[18 + 3 * j + 0] = __sinf(rx * fr);
    f[18 + 3 * j + 1] = __sinf(ry * fr);
    f[18 + 3 * j + 2] = __sinf(rz * fr);
    f[36 + 3 * j + 0] = __cosf(rx * fr);
    f[36 + 3 * j + 1] = __cosf(ry * fr);
    f[36 + 3 * j + 2] = __cosf(rz * fr);
    fr *= 2.0f;
  }
#pragma unroll
  for (int c = 54; c < 64; c++) f[c] = 0.0f;

  __align__(16) _Float16 hf[64];
#pragma unroll
  for (int c = 0; c < 64; c++) hf[c] = (_Float16)f[c];
  {
    h8* dstl = (h8*)&sFeat[tid][0];
    const h8* srcl = (const h8*)hf;
#pragma unroll
    for (int c = 0; c < 8; c++) dstl[c] = srcl[c];
  }
  __syncthreads();

  // ---- WMMA MLP phase: wave handles 2 tiles of 16 rows, in place in sFeat ----
  int lane = tid & 31, wid = tid >> 5;
  int m = lane & 15, hi = lane >> 4;
  float bv3 = b3[(m < 3) ? m : 2];
  bv3 = (m < 3) ? bv3 : 0.0f;

#pragma unroll
  for (int tt = 0; tt < 2; tt++) {
    int lrow0 = wid * 32 + tt * 16;
    _Float16* stg = &sFeat[lrow0][0];
    const _Float16* S = stg + m * FSTRIDE;
    // layer 0 (feature rows, in-place)
    v16h a0 = ldfrag(S + hi * 8, S + hi * 8 + 16);
    v16h a1 = ldfrag(S + 32 + hi * 8, S + 32 + hi * 8 + 16);
    mlp_layer(a0, a1, &sWT[0][0][0], b0, stg, lane);
    // layer 1
    a0 = ldfrag(S + hi * 8, S + hi * 8 + 16);
    a1 = ldfrag(S + 32 + hi * 8, S + 32 + hi * 8 + 16);
    mlp_layer(a0, a1, &sWT[1][0][0], b1, stg, lane);
    // layer 2
    a0 = ldfrag(S + hi * 8, S + hi * 8 + 16);
    a1 = ldfrag(S + 32 + hi * 8, S + 32 + hi * 8 + 16);
    mlp_layer(a0, a1, &sWT[2][0][0], b2, stg, lane);
    // output layer: single N-tile (3 real cols), sigmoid, masked plane store
    a0 = ldfrag(S + hi * 8, S + hi * 8 + 16);
    a1 = ldfrag(S + 32 + hi * 8, S + 32 + hi * 8 + 16);
    v8f c = {bv3, bv3, bv3, bv3, bv3, bv3, bv3, bv3};
    const _Float16* p0w = &sW3T[0][0] + m * 64 + hi * 8;
    v16h bf0 = ldfrag(p0w, p0w + 16);
    c = __builtin_amdgcn_wmma_f32_16x16x32_f16(false, a0, false, bf0, (short)0, c,
                                               false, false);
    const _Float16* p1w = p0w + 32;
    v16h bf1 = ldfrag(p1w, p1w + 16);
    c = __builtin_amdgcn_wmma_f32_16x16x32_f16(false, a1, false, bf1, (short)0, c,
                                               false, false);
    if (m < 3) {
#pragma unroll
      for (int r2 = 0; r2 < 8; r2++) {
        int lr = lrow0 + r2 + hi * 8;
        int mm = blockIdx.x * 256 + lr;
        float v = sMask[lr] ? (1.0f / (1.0f + __expf(-c[r2]))) : 0.0f;
        rgb[(size_t)m * NPTS + mm] = v;
      }
    }
  }
}

// ---------------------------------------------------------------------------
// Kernel 3: per-ray compositing (alpha + rgb planes are sample-major)
// ---------------------------------------------------------------------------
__global__ __launch_bounds__(256) void composite_kernel(
    const float* __restrict__ alpha, const float* __restrict__ rgb,
    float* __restrict__ out) {
  int r = blockIdx.x * 256 + threadIdx.x;
  if (r >= N_RAYS) return;
  const float* rgb0 = rgb;
  const float* rgb1 = rgb + NPTS;
  const float* rgb2 = rgb + 2 * (size_t)NPTS;
  float T = 1.0f, acc0 = 0.f, acc1 = 0.f, acc2 = 0.f;
  for (int s = 0; s < N_SAMP; s++) {
    int p = s * N_RAYS + r;
    float a = alpha[p];
    float w = a * T;
    acc0 += w * rgb0[p];
    acc1 += w * rgb1[p];
    acc2 += w * rgb2[p];
    T *= fmaxf(1.0f - a, 1e-10f);
  }
  out[r * 3 + 0] = acc0 + T * 1.0f;   // BG = 1.0
  out[r * 3 + 1] = acc1 + T * 1.0f;
  out[r * 3 + 2] = acc2 + T * 1.0f;
}

// ---------------------------------------------------------------------------
extern "C" void kernel_launch(void* const* d_in, const int* in_sizes, int n_in,
                              void* d_out, int out_size, void* d_ws,
                              size_t ws_size, hipStream_t stream) {
  const float* rays_o   = (const float*)d_in[0];
  const float* rays_d   = (const float*)d_in[1];
  const float* viewdirs = (const float*)d_in[2];
  const float* grid     = (const float*)d_in[3];
  const float* sobel    = (const float*)d_in[4];
  const float* w0 = (const float*)d_in[5];  const float* b0 = (const float*)d_in[6];
  const float* w1 = (const float*)d_in[7];  const float* b1 = (const float*)d_in[8];
  const float* w2 = (const float*)d_in[9];  const float* b2 = (const float*)d_in[10];
  const float* w3 = (const float*)d_in[11]; const float* b3 = (const float*)d_in[12];

  char* base = (char*)d_ws;
  float* normals = (float*)base; base += (size_t)3 * RES3 * 4;   // 25.2 MB
  float* alpha   = (float*)base; base += (size_t)NPTS * 4;       //  3.6 MB
  float* rgb     = (float*)base;                                  // 10.9 MB (3 planes)

  sobel_normals_kernel<<<RES3 / 256, 256, 0, stream>>>(grid, sobel, normals);
  fused_sample_mlp_kernel<<<NPTS / 256, 256, 0, stream>>>(
      rays_o, rays_d, viewdirs, grid, normals, w0, b0, w1, b1, w2, b2, w3, b3,
      alpha, rgb);
  composite_kernel<<<(N_RAYS + 255) / 256, 256, 0, stream>>>(alpha, rgb,
                                                             (float*)d_out);
}
